// MCAN_CB_87909390614893
// MI455X (gfx1250) — compile-verified, matrix operand order checked by reading, not run on hardware
//
#include <hip/hip_runtime.h>
#include <hip/hip_bf16.h>
#include <cstdint>
#include <cstddef>

// ---- problem dims ----
#define B_    128
#define LQ_   128
#define LA_   512
#define E_    300
#define EP_   320      // E padded to multiple of 32 (bf16 WMMA K-step)
#define H_    512
#define G_    2048     // 4*H
#define DP_   320      // D_IN=312 padded
#define DEC_  200
#define DECP_ 256      // padded so N-tiles divisible by 4 (wave does 4 N tiles)
#define NEG_  -1e9f
#define MQ_   (B_*LQ_)     // 16384
#define MA_   (B_*LA_)     // 65536
#define MT_   (MQ_+MA_)    // 81920

typedef __bf16 bf16;
typedef __attribute__((ext_vector_type(16))) __bf16 v16bf;
typedef __attribute__((ext_vector_type(8)))  __bf16 v8bf;
typedef __attribute__((ext_vector_type(8)))  float  v8f;
typedef unsigned int u32x4 __attribute__((ext_vector_type(4)));
typedef int          i32x4 __attribute__((ext_vector_type(4)));
typedef int          i32x8 __attribute__((ext_vector_type(8)));

#if __has_include(<hip/amd_detail/amd_gfx1250_TDM.h>)
#define TDM_6ARG 1
#endif

__device__ __forceinline__ float bf2f(bf16 v) {
  unsigned short u = __builtin_bit_cast(unsigned short, v);
  unsigned int x = ((unsigned int)u) << 16;
  return __builtin_bit_cast(float, x);
}
__device__ __forceinline__ bf16 f2bf(float f) {
  unsigned int x = __builtin_bit_cast(unsigned int, f);
  unsigned int r = x + 0x7fffu + ((x >> 16) & 1u);   // round-to-nearest-even
  unsigned short h = (unsigned short)(r >> 16);
  return __builtin_bit_cast(bf16, h);
}
__device__ __forceinline__ float wred_sum(float v) {
  #pragma unroll
  for (int i = 16; i > 0; i >>= 1) v += __shfl_xor(v, i, 32);
  return v;
}

// ISA 7.12.2 bf16 A/B fragment: low lanes hold K 0..7 & 16..23, high lanes K 8..15 & 24..31.
// p points at rowBase + kb + (lane<16 ? 0 : 8).
__device__ __forceinline__ v16bf load_frag16(const bf16* p) {
  v8bf lo = *(const v8bf*)(p);
  v8bf hi = *(const v8bf*)(p + 16);
  return __builtin_shufflevector(lo, hi, 0,1,2,3,4,5,6,7,8,9,10,11,12,13,14,15);
}

// ================= GEMM: A row-major [M,K], B given as rows of B^T (Wrow[n][k]) =============
// Each wave computes a 16x64 C strip (4 N tiles): one A fragment reused across 4 B fragments.
__global__ void gemm_rrT(const bf16* __restrict__ A, long long strideA, int lda,
                         const bf16* __restrict__ Bw, long long strideB, int ldb,
                         int M, int K,
                         float* Cf, bf16* Cb, long long strideC, int ldc,
                         const float* __restrict__ bias, int relu,
                         const int* __restrict__ a_len, int maskTok)
{
  int lane = threadIdx.x & 31, wid = threadIdx.x >> 5;
  int tn4 = blockIdx.x;               // group of 4 N tiles
  int tm = blockIdx.y * 4 + wid;
  if (tm * 16 >= M) return;
  int bz = blockIdx.z;
  const bf16* Ab = A + (size_t)bz * strideA;
  const bf16* Bb = Bw + (size_t)bz * strideB;
  int l15 = lane & 15;
  int off = (lane < 16) ? 0 : 8;
  const bf16* arow  = Ab + (size_t)(tm * 16 + l15) * lda + off;
  const bf16* brow0 = Bb + (size_t)(tn4 * 64 + l15) * ldb + off;
  v8f acc[4];
  #pragma unroll
  for (int j = 0; j < 4; ++j) acc[j] = (v8f){0.f,0.f,0.f,0.f,0.f,0.f,0.f,0.f};
  for (int kb = 0; kb < K; kb += 32) {
    __builtin_prefetch(arow + kb + 128, 0, 3);   // WGP-scope global_prefetch_b8
    v16bf fa = load_frag16(arow + kb);
    #pragma unroll
    for (int j = 0; j < 4; ++j) {
      v16bf fb = load_frag16(brow0 + (size_t)j * 16 * ldb + kb);
      acc[j] = __builtin_amdgcn_wmma_f32_16x16x32_bf16(false, fa, false, fb, (short)0, acc[j], false, false);
    }
  }
  int mhi = (lane >> 4) << 3;
  #pragma unroll
  for (int j = 0; j < 4; ++j) {
    int n = tn4 * 64 + j * 16 + l15;
    float bv = bias ? bias[n] : 0.f;
    #pragma unroll
    for (int r = 0; r < 8; ++r) {
      int m = tm * 16 + r + mhi;
      float v = acc[j][r] + bv;
      if (relu) v = fmaxf(v, 0.f);
      if (maskTok) { if ((m % maskTok) >= a_len[m / maskTok]) v = 0.f; }
      size_t ci = (size_t)bz * strideC + (size_t)m * ldc + n;
      if (Cf) Cf[ci] = v; else Cb[ci] = f2bf(v);
    }
  }
}

// ================= GEMM: A row-major [M,K], B row-major [K,N] (needs transpose) =============
// B tile staged through per-wave LDS; DS ops are in-order per wave, s_wait_dscnt makes it explicit.
__global__ void gemm_rr(const bf16* __restrict__ A, long long strideA, int lda,
                        const bf16* __restrict__ Bm, long long strideB, int ldb,
                        int M, int K,
                        float* __restrict__ C, long long strideC, int ldc)
{
  __shared__ bf16 lds[4][32][16];
  int lane = threadIdx.x & 31, wid = threadIdx.x >> 5;
  int tn = blockIdx.x;
  int tm = blockIdx.y * 4 + wid;
  if (tm * 16 >= M) return;
  int bz = blockIdx.z;
  const bf16* Ab = A + (size_t)bz * strideA;
  const bf16* Bb = Bm + (size_t)bz * strideB;
  int l15 = lane & 15;
  int off = (lane < 16) ? 0 : 8;
  const bf16* arow = Ab + (size_t)(tm * 16 + l15) * lda + off;
  v8f acc = {0.f,0.f,0.f,0.f,0.f,0.f,0.f,0.f};
  for (int kb = 0; kb < K; kb += 32) {
    const bf16* bsrc = Bb + (size_t)(kb + lane) * ldb + tn * 16;  // lane = K row of the tile
    *(v8bf*)&lds[wid][lane][0] = *(const v8bf*)bsrc;
    *(v8bf*)&lds[wid][lane][8] = *(const v8bf*)(bsrc + 8);
    asm volatile("s_wait_dscnt 0" ::: "memory");
    v16bf fa = load_frag16(arow + kb);
    v16bf fb;
    #pragma unroll
    for (int j = 0; j < 8; ++j) {
      fb[j]     = lds[wid][off + j][l15];
      fb[8 + j] = lds[wid][off + 16 + j][l15];
    }
    acc = __builtin_amdgcn_wmma_f32_16x16x32_bf16(false, fa, false, fb, (short)0, acc, false, false);
    asm volatile("s_wait_dscnt 0" ::: "memory");  // reads done before next-iter overwrite
  }
  int n = tn * 16 + l15;
  int mhi = (lane >> 4) << 3;
  #pragma unroll
  for (int r = 0; r < 8; ++r) {
    int m = tm * 16 + r + mhi;
    C[(size_t)bz * strideC + (size_t)m * ldc + n] = acc[r];
  }
}

// ================= fused LSTM step: TDM-staged Whh + recurrent WMMA + gates + pooling ========
// Block owns one hidden tile tn: its 64 Whh rows (4 gates x 16 rows x 512) = 64KB go to LDS
// via one Tensor Data Mover 3D-tile load (dim1-stride H*H hops between gate blocks).
__global__ void lstm_step(const float* __restrict__ xg_t, int tokStride,
                          const bf16* __restrict__ hin, bf16* __restrict__ hout,
                          const bf16* __restrict__ Whh, float* __restrict__ c,
                          float* __restrict__ psum, float* __restrict__ pmax,
                          const int* __restrict__ a_len, int t)
{
  __shared__ bf16 wtile[4 * 16 * H_];   // 64 KB of the WGP's 320 KB LDS
  int lane = threadIdx.x & 31, wid = threadIdx.x >> 5;
  int tn = blockIdx.x;               // hidden tile 0..31 (uniform per block)
  int tm = blockIdx.y * 4 + wid;     // batch tile 0..7

#if __has_builtin(__builtin_amdgcn_tensor_load_to_lds) && __has_builtin(__builtin_amdgcn_s_wait_tensorcnt)
  if (threadIdx.x == 0) {
    unsigned long long ga = (unsigned long long)(uintptr_t)(Whh + (size_t)tn * 16 * H_);
    unsigned ldsOff = (unsigned)(uintptr_t)(&wtile[0]);  // flat LDS addr low 32 bits == LDS offset
    // D# group0: count=1 | lds_addr | global_addr[56:0] | type=2
    u32x4 g0 = { 1u, ldsOff, (unsigned)ga, (unsigned)((ga >> 32) & 0x1FFFFFFu) | (2u << 30) };
    // D# group1: data_size=2B; tensor_dim0=512, tensor_dim1=2048; tile=512x16x4;
    // dim0_stride=512, dim1_stride=H*H=262144 (gate-block hop)
    i32x8 g1 = { 0x10000,
                 (int)(512u << 16),            // tensor_dim0[15:0] @ bits63:48
                 (int)(2048u << 16),           // dim0 hi=0, tensor_dim1[15:0] @ bits95:80
                 (int)(512u << 16),            // dim1 hi=0, tile_dim0 @ bits127:112
                 (int)(16u | (4u << 16)),      // tile_dim1=16, tile_dim2=4
                 512, 0,                       // tensor_dim0_stride = 512
                 4 };                          // tensor_dim1_stride = 0x40000 (hi bits)
    i32x4 g2 = { 4, 1, 0, 0 };                 // tensor_dim2=4, tensor_dim3=1
    i32x4 g3 = { 0, 0, 0, 0 };
#ifdef TDM_6ARG
    __builtin_amdgcn_tensor_load_to_lds(g0, g1, g2, g3, (i32x8){0,0,0,0,0,0,0,0}, 0);
#else
    __builtin_amdgcn_tensor_load_to_lds(g0, g1, g2, g3, 0);
#endif
    __builtin_amdgcn_s_wait_tensorcnt(0);
  }
  __syncthreads();
#else
  for (int i = threadIdx.x * 8; i < 4 * 16 * H_; i += blockDim.x * 8) {
    int g = i / (16 * H_); int rem = i % (16 * H_); int row = rem / H_; int col = rem % H_;
    *(v8bf*)&wtile[i] = *(const v8bf*)&Whh[((size_t)(g * H_ + tn * 16 + row)) * H_ + col];
  }
  __syncthreads();
#endif

  int l15 = lane & 15;
  int off = (lane < 16) ? 0 : 8;
  const bf16* arow = hin + (size_t)(tm * 16 + l15) * H_ + off;
  v8f acc[4];
  #pragma unroll
  for (int g = 0; g < 4; ++g) acc[g] = (v8f){0.f,0.f,0.f,0.f,0.f,0.f,0.f,0.f};
  for (int kb = 0; kb < H_; kb += 32) {
    v16bf fa = load_frag16(arow + kb);
    #pragma unroll
    for (int g = 0; g < 4; ++g) {
      const bf16* brow = &wtile[(size_t)(g * 16 + l15) * H_ + kb + off];  // ds_load_b128
      v16bf fb = load_frag16(brow);
      acc[g] = __builtin_amdgcn_wmma_f32_16x16x32_bf16(false, fa, false, fb, (short)0, acc[g], false, false);
    }
  }
  int hid = tn * 16 + l15;
  int mhi = (lane >> 4) << 3;
  #pragma unroll
  for (int r = 0; r < 8; ++r) {
    int b = tm * 16 + r + mhi;
    const float* xr = xg_t + (size_t)b * tokStride * G_;
    float gi = acc[0][r] + xr[hid];
    float gf = acc[1][r] + xr[H_ + hid];
    float gg = acc[2][r] + xr[2 * H_ + hid];
    float go = acc[3][r] + xr[3 * H_ + hid];
    float si = 1.f / (1.f + expf(-gi));
    float sf = 1.f / (1.f + expf(-gf));
    float so = 1.f / (1.f + expf(-go));
    float tg = tanhf(gg);
    size_t ci = (size_t)b * H_ + hid;
    float cn = sf * c[ci] + si * tg;
    float hn = so * tanhf(cn);
    c[ci] = cn;
    hout[ci] = f2bf(hn);
    bool ok = a_len ? (t < a_len[b]) : true;
    if (ok) { psum[ci] += hn; pmax[ci] = fmaxf(pmax[ci], hn); }
  }
}

// ================= elementwise / reduction kernels ============================================
__global__ void k_gather(const int* __restrict__ ids, const float* __restrict__ emb,
                         bf16* __restrict__ out, int tokensPerB, const int* __restrict__ a_len)
{
  int tok = blockIdx.x, e = threadIdx.x;   // block = EP_
  int id = ids[tok];
  bool masked = a_len && ((tok % tokensPerB) >= a_len[tok / tokensPerB]);
  float v = (e < E_ && !masked) ? emb[(size_t)id * E_ + e] : 0.f;
  out[(size_t)tok * EP_ + e] = f2bf(v);
}

__global__ void k_cvt_pad(const float* __restrict__ src, bf16* __restrict__ dst,
                          int Nsrc, int Ksrc, int Npad, int Kpad)
{
  long long idx = (long long)blockIdx.x * 256 + threadIdx.x;
  if (idx >= (long long)Npad * Kpad) return;
  int n = (int)(idx / Kpad), k = (int)(idx % Kpad);
  float v = (n < Nsrc && k < Ksrc) ? src[(size_t)n * Ksrc + k] : 0.f;
  dst[idx] = f2bf(v);
}

__global__ void k_pad_bias(const float* __restrict__ s1, const float* __restrict__ s2,
                           float* __restrict__ dst, int n, int npad)
{
  int i = blockIdx.x * 256 + threadIdx.x;
  if (i >= npad) return;
  dst[i] = (i < n) ? (s1[i] + (s2 ? s2[i] : 0.f)) : 0.f;
}

__global__ void k_s_stats(const float* __restrict__ s, const int* __restrict__ a_len,
                          float* smax2, float* smean2, float* smax1, float* smean1)
{
  int b = blockIdx.x, tid = threadIdx.x;
  int alen = a_len[b];
  const float* sb = s + (size_t)b * LQ_ * LA_;
  for (int q = tid; q < LQ_; q += 256) {
    const float* row = sb + (size_t)q * LA_;
    float mx = -3e38f, sm = 0.f;
    for (int a = 0; a < LA_; ++a) { float v = row[a]; sm += v; mx = fmaxf(mx, a < alen ? v : NEG_); }
    smax2[b * LQ_ + q] = mx; smean2[b * LQ_ + q] = sm / (float)alen;
  }
  for (int a = tid; a < LA_; a += 256) {
    float mx = -3e38f, sm = 0.f;
    for (int q = 0; q < LQ_; ++q) { float v = sb[(size_t)q * LA_ + a]; sm += v; mx = fmaxf(mx, v); }
    smax1[b * LA_ + a] = (a < alen) ? mx : NEG_;
    smean1[b * LA_ + a] = sm / (float)LQ_;
  }
}

// softmax(weights) @ emb -> attvec[b][which][0:300], attsum[b][which]
__global__ void k_attvec(const float* smax2, const float* smean2,
                         const float* smax1, const float* smean1,
                         const bf16* __restrict__ qe, const bf16* __restrict__ ae,
                         float* __restrict__ attvec, float* __restrict__ attsum)
{
  __shared__ float w[LA_];
  __shared__ float red[256];
  int which = blockIdx.x, b = blockIdx.y, tid = threadIdx.x;
  int L = which < 2 ? LQ_ : LA_;
  const float* sv = which == 0 ? smax2 + b * LQ_ :
                    which == 1 ? smean2 + b * LQ_ :
                    which == 2 ? smax1 + b * LA_ : smean1 + b * LA_;
  float mx = -3e38f;
  for (int i = tid; i < L; i += 256) mx = fmaxf(mx, sv[i]);
  red[tid] = mx; __syncthreads();
  for (int s2 = 128; s2 > 0; s2 >>= 1) { if (tid < s2) red[tid] = fmaxf(red[tid], red[tid + s2]); __syncthreads(); }
  mx = red[0]; __syncthreads();
  float sm = 0.f;
  for (int i = tid; i < L; i += 256) { float e = expf(sv[i] - mx); w[i] = e; sm += e; }
  red[tid] = sm; __syncthreads();
  for (int s2 = 128; s2 > 0; s2 >>= 1) { if (tid < s2) red[tid] += red[tid + s2]; __syncthreads(); }
  float inv = 1.f / red[0]; __syncthreads();
  const bf16* emb = which < 2 ? qe + (size_t)b * LQ_ * EP_ : ae + (size_t)b * LA_ * EP_;
  float psum = 0.f;
  for (int e = tid; e < E_; e += 256) {
    float acc = 0.f;
    for (int t = 0; t < L; ++t) acc += w[t] * bf2f(emb[(size_t)t * EP_ + e]);
    acc *= inv;
    attvec[((size_t)b * 4 + which) * EP_ + e] = acc;
    psum += acc;
  }
  red[tid] = psum; __syncthreads();
  for (int s2 = 128; s2 > 0; s2 >>= 1) { if (tid < s2) red[tid] += red[tid + s2]; __syncthreads(); }
  if (tid == 0) attsum[b * 4 + which] = red[0];
}

// generic row/col softmax -> bf16. maskMode: 0 none, 1 mask elements j>=alen to NEG, 2 masked row -> uniform
__global__ void k_softmax(const float* __restrict__ S, long long bStrideS, long long rStrideS, long long eStrideS,
                          bf16* __restrict__ O, long long bStrideO, long long rStrideO,
                          int n, const int* __restrict__ a_len, int maskMode)
{
  __shared__ float red[256];
  __shared__ float vals[LA_];
  int row = blockIdx.x, b = blockIdx.y, tid = threadIdx.x;
  const float* src = S + (size_t)b * bStrideS + (size_t)row * rStrideS;
  bf16* dst = O + (size_t)b * bStrideO + (size_t)row * rStrideO;
  int alen = a_len ? a_len[b] : 0;
  if (maskMode == 2 && row >= alen) {
    bf16 u = f2bf(1.f / (float)n);
    for (int j = tid; j < n; j += 256) dst[j] = u;
    return;
  }
  float mx = -3e38f;
  for (int j = tid; j < n; j += 256) {
    float v = src[(size_t)j * eStrideS];
    if (maskMode == 1 && j >= alen) v = NEG_;
    vals[j] = v; mx = fmaxf(mx, v);
  }
  red[tid] = mx; __syncthreads();
  for (int s2 = 128; s2 > 0; s2 >>= 1) { if (tid < s2) red[tid] = fmaxf(red[tid], red[tid + s2]); __syncthreads(); }
  mx = red[0]; __syncthreads();
  float sm = 0.f;
  for (int j = tid; j < n; j += 256) { float e = expf(vals[j] - mx); vals[j] = e; sm += e; }
  red[tid] = sm; __syncthreads();
  for (int s2 = 128; s2 > 0; s2 >>= 1) { if (tid < s2) red[tid] += red[tid + s2]; __syncthreads(); }
  float inv = 1.f / red[0]; __syncthreads();
  for (int j = tid; j < n; j += 256) dst[j] = f2bf(vals[j] * inv);
}

// per-token cast features + concat -> x2 bf16 [tok, DP_]; one wave per token
__global__ void k_feat(const bf16* __restrict__ emb, const float* __restrict__ align,
                       const float* __restrict__ intra,
                       const float* __restrict__ attvec, const float* __restrict__ attsum,
                       int whichBase, bf16* __restrict__ out, int tokensPerB, int Mtok)
{
  int wid = threadIdx.x >> 5, lane = threadIdx.x & 31;
  int tok = blockIdx.x * 8 + wid;
  if (tok >= Mtok) return;
  int b = tok / tokensPerB;
  const bf16* erow = emb + (size_t)tok * EP_;
  const float* arow = align + (size_t)tok * EP_;
  const float* irow = intra + (size_t)tok * EP_;
  const float* vmax  = attvec + ((size_t)b * 4 + whichBase) * EP_;
  const float* vmean = attvec + ((size_t)b * 4 + whichBase + 1) * EP_;
  float sb = 0, dmax = 0, dmean = 0, sal = 0, dal = 0, sit = 0, dit = 0;
  for (int e = lane; e < E_; e += 32) {
    float be = bf2f(erow[e]);
    sb += be;
    dmax  += vmax[e] * be;
    dmean += vmean[e] * be;
    float al = arow[e]; sal += al; dal += al * be;
    float it = irow[e]; sit += it; dit += it * be;
  }
  sb = wred_sum(sb); dmax = wred_sum(dmax); dmean = wred_sum(dmean);
  sal = wred_sum(sal); dal = wred_sum(dal); sit = wred_sum(sit); dit = wred_sum(dit);
  bf16* orow = out + (size_t)tok * DP_;
  for (int e = lane; e < E_; e += 32) orow[e] = erow[e];
  if (lane == 0) {
    float sa0 = attsum[b * 4 + whichBase], sa1 = attsum[b * 4 + whichBase + 1];
    float f[12] = { sa0 + sb, dmax,  sa0 - sb,
                    sa1 + sb, dmean, sa1 - sb,
                    sal + sb, dal,   sal - sb,
                    sit + sb, dit,   sit - sb };
    for (int i = 0; i < 12; ++i) orow[E_ + i] = f2bf(f[i]);
    for (int i = E_ + 12; i < DP_; ++i) orow[i] = f2bf(0.f);
  }
}

__global__ void k_init_state(bf16* h, float* c, float* qs, float* qm, float* as, float* am)
{
  int i = blockIdx.x * 256 + threadIdx.x;
  if (i >= B_ * H_) return;
  h[i] = f2bf(0.f); c[i] = 0.f;
  qs[i] = 0.f; qm[i] = -3e38f; as[i] = 0.f; am[i] = -1e30f;
}

__global__ void k_head(const float* qsum, const float* qmax, const float* asum, const float* amax,
                       const int* a_len, bf16* xcat)
{
  int idx = blockIdx.x * 256 + threadIdx.x;
  if (idx >= B_ * 1024) return;
  int b = idx >> 10, k = idx & 1023;
  float alen = (float)a_len[b];
  float hq = (k < H_) ? qsum[(size_t)b * H_ + k] / (float)LQ_ : qmax[(size_t)b * H_ + (k - H_)];
  float ha = (k < H_) ? asum[(size_t)b * H_ + k] / alen       : amax[(size_t)b * H_ + (k - H_)];
  bf16* x = xcat + (size_t)b * 4096;
  x[k] = f2bf(hq); x[1024 + k] = f2bf(ha);
  x[2048 + k] = f2bf(ha * hq); x[3072 + k] = f2bf(hq - ha);
}

__global__ void k_highway(const float* __restrict__ xin, float* __restrict__ xout,
                          const float* __restrict__ gW, const float* __restrict__ gB,
                          const float* __restrict__ nW, const float* __restrict__ nB)
{
  __shared__ float xr[DEC_];
  int b = blockIdx.x, tid = threadIdx.x;
  for (int j = tid; j < DEC_; j += 256) xr[j] = xin[(size_t)b * DECP_ + j];
  __syncthreads();
  for (int j = tid; j < DEC_; j += 256) {
    float g = gB[j], nl = nB[j];
    for (int k = 0; k < DEC_; ++k) { float xv = xr[k]; g += gW[(size_t)j * DEC_ + k] * xv; nl += nW[(size_t)j * DEC_ + k] * xv; }
    g = 1.f / (1.f + expf(-g));
    nl = fmaxf(nl, 0.f);
    xout[(size_t)b * DECP_ + j] = g * nl + (1.f - g) * xr[j];
  }
}

__global__ void k_final(const float* __restrict__ x, const float* __restrict__ d2W,
                        const float* __restrict__ d2b, float* __restrict__ out)
{
  int b = blockIdx.x, lane = threadIdx.x;   // block 32
  float l0 = 0, l1 = 0;
  for (int k = lane; k < DEC_; k += 32) {
    float xv = x[(size_t)b * DECP_ + k];
    l0 += d2W[k] * xv; l1 += d2W[DEC_ + k] * xv;
  }
  l0 = wred_sum(l0); l1 = wred_sum(l1);
  if (lane == 0) {
    l0 += d2b[0]; l1 += d2b[1];
    float m = fmaxf(l0, l1);
    float e0 = expf(l0 - m), e1 = expf(l1 - m);
    out[b] = e0 / (e0 + e1);
  }
}

// =============================================================================================
extern "C" void kernel_launch(void* const* d_in, const int* in_sizes, int n_in,
                              void* d_out, int out_size, void* d_ws, size_t ws_size,
                              hipStream_t stream)
{
  (void)in_sizes; (void)n_in; (void)out_size; (void)ws_size;
  const int*   q_ids   = (const int*)d_in[0];
  const int*   a_ids   = (const int*)d_in[1];
  const int*   a_len   = (const int*)d_in[2];
  const float* emb     = (const float*)d_in[3];
  const float* co_W    = (const float*)d_in[4];
  const float* co_b    = (const float*)d_in[5];
  const float* iq_W    = (const float*)d_in[6];
  const float* iq_b    = (const float*)d_in[7];
  const float* ia_W    = (const float*)d_in[8];
  const float* ia_b    = (const float*)d_in[9];
  const float* Wih     = (const float*)d_in[10];
  const float* Whh     = (const float*)d_in[11];
  const float* bih     = (const float*)d_in[12];
  const float* bhh     = (const float*)d_in[13];
  const float* d1_W    = (const float*)d_in[14];
  const float* d1_b    = (const float*)d_in[15];
  const float* hw_nl_W = (const float*)d_in[16];
  const float* hw_nl_b = (const float*)d_in[17];
  const float* hw_g_W  = (const float*)d_in[18];
  const float* hw_g_b  = (const float*)d_in[19];
  const float* d2_W    = (const float*)d_in[20];
  const float* d2_b    = (const float*)d_in[21];
  float* out = (float*)d_out;
  char* ws = (char*)d_ws;

  size_t off = 0;
  auto take = [&](size_t bytes) { size_t r = off; off = (off + bytes + 255) & ~(size_t)255; return r; };
  // --- persistent scratch ---
  size_t oWco  = take((size_t)EP_*EP_*2), oWiq = take((size_t)EP_*EP_*2), oWia = take((size_t)EP_*EP_*2);
  size_t oBco  = take(EP_*4), oBiq = take(EP_*4), oBia = take(EP_*4);
  size_t oWih  = take((size_t)G_*DP_*2);
  size_t oWhh  = take((size_t)G_*H_*2);
  size_t oBsum = take(G_*4);
  size_t oWd1  = take((size_t)DECP_*4096*2);
  size_t oBd1  = take(DECP_*4);
  size_t oQe   = take((size_t)MQ_*EP_*2);
  size_t oAe   = take((size_t)MA_*EP_*2);
  size_t oQatt = take((size_t)MQ_*EP_*2);
  size_t oAatt = take((size_t)MA_*EP_*2);
  size_t oQi   = take((size_t)MQ_*EP_*2);
  size_t oAi   = take((size_t)MA_*EP_*2);
  size_t oSmax2 = take(B_*LQ_*4), oSmean2 = take(B_*LQ_*4);
  size_t oSmax1 = take(B_*LA_*4), oSmean1 = take(B_*LA_*4);
  size_t oAttvec = take((size_t)B_*4*EP_*4), oAttsum = take(B_*4*4);
  size_t oQ2A2 = take((size_t)MT_*DP_*2);
  size_t oH0 = take((size_t)B_*H_*2), oH1 = take((size_t)B_*H_*2);
  size_t oC  = take((size_t)B_*H_*4);
  size_t oQsum = take((size_t)B_*H_*4), oQmax = take((size_t)B_*H_*4);
  size_t oAsum = take((size_t)B_*H_*4), oAmax = take((size_t)B_*H_*4);
  size_t oXcat = take((size_t)B_*4096*2);
  size_t oXd0 = take(B_*DECP_*4), oXd1 = take(B_*DECP_*4);
  // --- union region: attention scratch is dead before xg is written, so xg aliases it ---
  size_t ub = off, u = off;
  auto takeU = [&](size_t bytes) { size_t r = u; u = (u + bytes + 255) & ~(size_t)255; return r; };
  size_t oS   = takeU((size_t)B_*LQ_*LA_*4);
  size_t oSq  = takeU((size_t)B_*LQ_*LQ_*4);
  size_t oSa  = takeU((size_t)B_*LA_*LA_*4);
  size_t oAq  = takeU((size_t)B_*LQ_*LA_*2);
  size_t oAaT = takeU((size_t)B_*LA_*LQ_*2);
  size_t oPq  = takeU((size_t)B_*LQ_*LQ_*2);
  size_t oPa  = takeU((size_t)B_*LA_*LA_*2);
  size_t oQal = takeU((size_t)MQ_*EP_*4);
  size_t oAal = takeU((size_t)MA_*EP_*4);
  size_t oQin = takeU((size_t)MQ_*EP_*4);
  size_t oAin = takeU((size_t)MA_*EP_*4);
  size_t oXg  = ub;   // f32 [MT_, G_] aliases all of the above

  auto Pb = [&](size_t o2) { return (bf16*)(ws + o2); };
  auto Pf = [&](size_t o2) { return (float*)(ws + o2); };

  auto cvt = [&](const float* src, size_t dst, int Ns, int Ks, int Np, int Kp) {
    long long tot = (long long)Np * Kp;
    k_cvt_pad<<<dim3((unsigned)((tot + 255) / 256)), 256, 0, stream>>>(src, Pb(dst), Ns, Ks, Np, Kp);
  };
  auto rrT = [&](size_t A, long long sA, int lda, size_t Bw, long long sB, int ldb,
                 int M, int N, int K, float* Cf, bf16* Cb, long long sC, int ldc,
                 const float* bias, int relu, const int* alen, int maskTok, int batches) {
    dim3 grid(N / 64, (unsigned)((M / 16 + 3) / 4), batches);
    gemm_rrT<<<grid, 128, 0, stream>>>(Pb(A), sA, lda, Pb(Bw), sB, ldb, M, K,
                                       Cf, Cb, sC, ldc, bias, relu, alen, maskTok);
  };
  auto rr = [&](size_t A, long long sA, int lda, size_t Bm, long long sB, int ldb,
                int M, int N, int K, size_t C, long long sC, int ldc, int batches) {
    dim3 grid(N / 16, (unsigned)((M / 16 + 3) / 4), batches);
    gemm_rr<<<grid, 128, 0, stream>>>(Pb(A), sA, lda, Pb(Bm), sB, ldb, M, K, Pf(C), sC, ldc);
  };

  // 1. param conversion (f32 -> zero-padded bf16)
  cvt(co_W, oWco, E_, E_, EP_, EP_);
  cvt(iq_W, oWiq, E_, E_, EP_, EP_);
  cvt(ia_W, oWia, E_, E_, EP_, EP_);
  cvt(Wih,  oWih, G_, 312, G_, DP_);
  cvt(Whh,  oWhh, G_, H_, G_, H_);
  cvt(d1_W, oWd1, DEC_, 4096, DECP_, 4096);
  k_pad_bias<<<2, 256, 0, stream>>>(co_b, nullptr, Pf(oBco), E_, EP_);
  k_pad_bias<<<2, 256, 0, stream>>>(iq_b, nullptr, Pf(oBiq), E_, EP_);
  k_pad_bias<<<2, 256, 0, stream>>>(ia_b, nullptr, Pf(oBia), E_, EP_);
  k_pad_bias<<<8, 256, 0, stream>>>(bih, bhh, Pf(oBsum), G_, G_);
  k_pad_bias<<<1, 256, 0, stream>>>(d1_b, nullptr, Pf(oBd1), DEC_, DECP_);

  // 2. embedding gathers (a masked to zero past a_len)
  k_gather<<<MQ_, EP_, 0, stream>>>(q_ids, emb, Pb(oQe), LQ_, nullptr);
  k_gather<<<MA_, EP_, 0, stream>>>(a_ids, emb, Pb(oAe), LA_, a_len);

  // 3. projections (bf16 WMMA, bias+relu; a_att additionally row-masked)
  rrT(oQe, 0, EP_, oWco, 0, EP_, MQ_, EP_, EP_, nullptr, Pb(oQatt), 0, EP_, Pf(oBco), 1, nullptr, 0, 1);
  rrT(oAe, 0, EP_, oWco, 0, EP_, MA_, EP_, EP_, nullptr, Pb(oAatt), 0, EP_, Pf(oBco), 1, a_len, LA_, 1);
  rrT(oQe, 0, EP_, oWiq, 0, EP_, MQ_, EP_, EP_, nullptr, Pb(oQi),   0, EP_, Pf(oBiq), 1, nullptr, 0, 1);
  rrT(oAe, 0, EP_, oWia, 0, EP_, MA_, EP_, EP_, nullptr, Pb(oAi),   0, EP_, Pf(oBia), 1, nullptr, 0, 1);

  // 4. score GEMMs (batched over B; both operands K-contiguous)
  rrT(oQatt, (long long)LQ_*EP_, EP_, oAatt, (long long)LA_*EP_, EP_,
      LQ_, LA_, EP_, Pf(oS), nullptr, (long long)LQ_*LA_, LA_, nullptr, 0, nullptr, 0, B_);
  rrT(oQi, (long long)LQ_*EP_, EP_, oQi, (long long)LQ_*EP_, EP_,
      LQ_, LQ_, EP_, Pf(oSq), nullptr, (long long)LQ_*LQ_, LQ_, nullptr, 0, nullptr, 0, B_);
  rrT(oAi, (long long)LA_*EP_, EP_, oAi, (long long)LA_*EP_, EP_,
      LA_, LA_, EP_, Pf(oSa), nullptr, (long long)LA_*LA_, LA_, nullptr, 0, nullptr, 0, B_);

  // 5. score stats + pooled attention vectors
  k_s_stats<<<B_, 256, 0, stream>>>(Pf(oS), a_len, Pf(oSmax2), Pf(oSmean2), Pf(oSmax1), Pf(oSmean1));
  k_attvec<<<dim3(4, B_), 256, 0, stream>>>(Pf(oSmax2), Pf(oSmean2), Pf(oSmax1), Pf(oSmean1),
                                            Pb(oQe), Pb(oAe), Pf(oAttvec), Pf(oAttsum));

  // 6. softmaxes -> bf16 probability matrices
  k_softmax<<<dim3(LQ_, B_), 256, 0, stream>>>(Pf(oS), (long long)LQ_*LA_, LA_, 1,
      Pb(oAq), (long long)LQ_*LA_, LA_, LA_, a_len, 1);                               // A_q rows
  k_softmax<<<dim3(LA_, B_), 256, 0, stream>>>(Pf(oS), (long long)LQ_*LA_, 1, LA_,
      Pb(oAaT), (long long)LA_*LQ_, LQ_, LQ_, nullptr, 0);                            // A_a cols (transposed out)
  k_softmax<<<dim3(LQ_, B_), 256, 0, stream>>>(Pf(oSq), (long long)LQ_*LQ_, LQ_, 1,
      Pb(oPq), (long long)LQ_*LQ_, LQ_, LQ_, nullptr, 0);
  k_softmax<<<dim3(LA_, B_), 256, 0, stream>>>(Pf(oSa), (long long)LA_*LA_, LA_, 1,
      Pb(oPa), (long long)LA_*LA_, LA_, LA_, a_len, 2);

  // 7. alignment GEMMs (B row-major -> LDS-transposed fragments)
  rr(oAq,  (long long)LQ_*LA_, LA_, oAe, (long long)LA_*EP_, EP_, LQ_, EP_, LA_, oQal, (long long)LQ_*EP_, EP_, B_);
  rr(oAaT, (long long)LA_*LQ_, LQ_, oQe, (long long)LQ_*EP_, EP_, LA_, EP_, LQ_, oAal, (long long)LA_*EP_, EP_, B_);
  rr(oPq,  (long long)LQ_*LQ_, LQ_, oQe, (long long)LQ_*EP_, EP_, LQ_, EP_, LQ_, oQin, (long long)LQ_*EP_, EP_, B_);
  rr(oPa,  (long long)LA_*LA_, LA_, oAe, (long long)LA_*EP_, EP_, LA_, EP_, LA_, oAin, (long long)LA_*EP_, EP_, B_);

  // 8. cast features + concat -> x2
  k_feat<<<MQ_/8, 256, 0, stream>>>(Pb(oQe), Pf(oQal), Pf(oQin), Pf(oAttvec), Pf(oAttsum), 0,
                                    Pb(oQ2A2), LQ_, MQ_);
  k_feat<<<MA_/8, 256, 0, stream>>>(Pb(oAe), Pf(oAal), Pf(oAin), Pf(oAttvec), Pf(oAttsum), 2,
                                    Pb(oQ2A2) + (size_t)MQ_*DP_, LA_, MA_);

  // 9. xg = x2 @ Wih^T + (bih+bhh)  — one 81920x2048x320 WMMA GEMM (aliases attention scratch)
  rrT(oQ2A2, 0, DP_, oWih, 0, DP_, MT_, G_, DP_, Pf(oXg), nullptr, 0, G_, Pf(oBsum), 0, nullptr, 0, 1);

  // 10. LSTM recurrence: 640 fused step kernels, double-buffered h, TDM-staged Whh
  k_init_state<<<(B_*H_ + 255)/256, 256, 0, stream>>>(Pb(oH0), Pf(oC), Pf(oQsum), Pf(oQmax), Pf(oAsum), Pf(oAmax));
  bf16* hA = Pb(oH0); bf16* hB = Pb(oH1);
  const float* xgq = Pf(oXg);
  const float* xga = Pf(oXg) + (size_t)MQ_ * G_;
  for (int t = 0; t < LQ_; ++t) {
    lstm_step<<<dim3(H_/16, 2), 128, 0, stream>>>(xgq + (size_t)t * G_, LQ_, hA, hB, Pb(oWhh),
        Pf(oC), Pf(oQsum), Pf(oQmax), nullptr, t);
    bf16* tmp = hA; hA = hB; hB = tmp;
  }
  for (int t = 0; t < LA_; ++t) {
    lstm_step<<<dim3(H_/16, 2), 128, 0, stream>>>(xga + (size_t)t * G_, LA_, hA, hB, Pb(oWhh),
        Pf(oC), Pf(oAsum), Pf(oAmax), a_len, t);
    bf16* tmp = hA; hA = hB; hB = tmp;
  }

  // 11. head: pooled concat -> d1 (WMMA) -> highway x2 -> softmax prob[:,0]
  k_head<<<(B_*1024 + 255)/256, 256, 0, stream>>>(Pf(oQsum), Pf(oQmax), Pf(oAsum), Pf(oAmax), a_len, Pb(oXcat));
  rrT(oXcat, 0, 4096, oWd1, 0, 4096, B_, DECP_, 4096, Pf(oXd0), nullptr, 0, DECP_, Pf(oBd1), 1, nullptr, 0, 1);
  k_highway<<<B_, 256, 0, stream>>>(Pf(oXd0), Pf(oXd1), hw_g_W, hw_g_b, hw_nl_W, hw_nl_b);
  k_highway<<<B_, 256, 0, stream>>>(Pf(oXd1), Pf(oXd0), hw_g_W + DEC_*DEC_, hw_g_b + DEC_,
                                    hw_nl_W + DEC_*DEC_, hw_nl_b + DEC_);
  k_final<<<B_, 32, 0, stream>>>(Pf(oXd0), d2_W, d2_b, out);
}
